// GATLayer_42649025249712
// MI455X (gfx1250) — matrix operand order repaired and checked by reading, not compiled, not run on hardware
//
#include <hip/hip_runtime.h>
#include <hip/hip_bf16.h>

// ---------------------------------------------------------------------------
// GAT layer for MI455X (gfx1250, wave32, WMMA + TDM).
//
//   K0: Wt[col][k] = bf16(W[k][col])                  (one-off transpose)
//   K1: proj GEMM X@W via bf16 WMMA; epilogue stores VT[b,h,d][m] (bf16,
//       m-major) -- each lane's 8 C-rows are 8 consecutive m -> 16B store.
//   K2: s/t = <VT, fc1/fc2> (coalesced over m)
//   K3: fused masked softmax attention:
//       - adjacency 16x32 tile DMA'd by the Tensor Data Mover into LDS
//         (padded [16][33] via D# pad feature), double-buffered,
//         s_wait_tensorcnt + barriers; tile shared by all 8 heads.
//       - P = A ? exp(leaky(s_n+t_m)) : 0 built in WMMA A-register layout
//       - O += P @ V (two v_wmma_f32_16x16x32_bf16), Z += P.1
//       - out = relu(O/Z)
// All WMMA operands are contiguous 32B loads (b128-pairs) thanks to the
// transposed layouts. Workspace: VT 4MB + Wt 128KB + s,t 512KB ~= 4.85MB.
// ---------------------------------------------------------------------------

typedef __attribute__((ext_vector_type(16))) __bf16 v16bf;
typedef __attribute__((ext_vector_type(8)))  __bf16 v8bf;
typedef __attribute__((ext_vector_type(8)))  float  v8f;
typedef __attribute__((ext_vector_type(4)))  float  f4;
typedef __attribute__((ext_vector_type(4)))  unsigned int u32x4;
typedef __attribute__((ext_vector_type(4)))  int     i32x4;
typedef __attribute__((ext_vector_type(8)))  int     i32x8;

constexpr int Bn = 8, Nn = 1024, IND = 256, OUTD = 256, Hh = 8, Dd = 32;

// A-matrix 16x32 (16-bit): K = half*16 + i (contiguous, matching B).
__device__ __forceinline__ int kmapA(int i, int half) {
  return i + half * 16;
}
// B-matrix 32x16 (16-bit): column N = lane%16, K = half*16 + i (contiguous!).

// ---------------------------------------------------------------------------
// K0: transpose-convert W -> Wt[col][k] bf16 so K1's B operand is contiguous.
// ---------------------------------------------------------------------------
__global__ __launch_bounds__(256) void gat_wt(const float* __restrict__ W,
                                              __bf16* __restrict__ Wt) {
  const int tid = blockIdx.x * 256 + threadIdx.x;   // 65536
  const int k = tid & 255, col = tid >> 8;
  Wt[(size_t)col * IND + k] = (__bf16)W[(size_t)k * OUTD + col];
}

// ---------------------------------------------------------------------------
// K1: [8192,256]x[256,256] bf16 WMMA GEMM -> VT[b,h,d][m] (m-major bf16).
// One 16x16 tile per wave; A loads = 4 x float4, B loads = 1 x v16bf.
// ---------------------------------------------------------------------------
__global__ __launch_bounds__(256) void gat_proj_wmma(
    const float* __restrict__ X, const __bf16* __restrict__ Wt,
    __bf16* __restrict__ VT) {
  const int lane = threadIdx.x & 31;
  const int wid  = threadIdx.x >> 5;
  const int job  = blockIdx.x * 8 + wid;   // 0..8191
  const int mt   = job >> 4;               // 512 row tiles
  const int nt   = job & 15;               // 16 col tiles
  const int r    = lane & 15;
  const int half = lane >> 4;

  v8f acc = {};
  for (int k0 = 0; k0 < IND; k0 += 32) {
    // A: one contiguous 16-float run -> 4 x b128
    const f4* xp = (const f4*)(X + (size_t)(mt * 16 + r) * IND + k0 + half * 16);
    const f4 x0 = xp[0], x1 = xp[1], x2 = xp[2], x3 = xp[3];
    v16bf a;
#pragma unroll
    for (int i = 0; i < 4; ++i) {
      a[i]      = (__bf16)x0[i];
      a[i + 4]  = (__bf16)x1[i];
      a[i + 8]  = (__bf16)x2[i];
      a[i + 12] = (__bf16)x3[i];
    }
    // B: 16 contiguous bf16 from transposed weights -> 2 x b128
    const v16bf b =
        *(const v16bf*)(Wt + (size_t)(nt * 16 + r) * IND + k0 + half * 16);
    acc = __builtin_amdgcn_wmma_f32_16x16x32_bf16(false, a, false, b,
                                                  (short)0, acc, false, false);
  }

  // Epilogue directly in V^T layout: lane holds column (nt*16+r) for rows
  // mt*16 + half*8 + j, j=0..7 -> 8 consecutive m -> one 16B store.
  const int colg = nt * 16 + r;
  const int h = colg >> 5, dl = colg & 31;
  const int row0 = mt * 16 + half * 8;
  const int b = row0 >> 10, n = row0 & (Nn - 1);
  v8bf st;
#pragma unroll
  for (int j = 0; j < 8; ++j) st[j] = (__bf16)acc[j];
  *(v8bf*)(VT + (((size_t)b * Hh + h) * Dd + dl) * Nn + n) = st;
}

// ---------------------------------------------------------------------------
// K2: s/t score halves from VT, coalesced over m. Block = (b,h).
// ---------------------------------------------------------------------------
__global__ __launch_bounds__(256) void gat_scores(
    const __bf16* __restrict__ VT, const float* __restrict__ fc1,
    const float* __restrict__ fc2, float* __restrict__ s_ws,
    float* __restrict__ t_ws) {
  const int bh = blockIdx.x;                 // 64 blocks
  const __bf16* base = VT + (size_t)bh * Dd * Nn;
  const int h = bh & (Hh - 1);
  for (int n = threadIdx.x; n < Nn; n += 256) {
    float s = 0.f, t = 0.f;
#pragma unroll
    for (int d = 0; d < Dd; ++d) {
      const float x = (float)base[(size_t)d * Nn + n];   // coalesced per d
      s += x * fc1[h * Dd + d];
      t += x * fc2[h * Dd + d];
    }
    s_ws[(size_t)bh * Nn + n] = s;
    t_ws[(size_t)bh * Nn + n] = t;
  }
}

// ---------------------------------------------------------------------------
// TDM: DMA one 16x32 f32 adjacency tile into LDS with 1-dword row padding
// (pad_interval=32 dwords, pad_amount=1 -> [16][33] layout, conflict-free).
// ---------------------------------------------------------------------------
__device__ __forceinline__ void tdm_load_a_tile(const float* gsrc,
                                                unsigned lds_off) {
  const unsigned long long ga = (unsigned long long)(uintptr_t)gsrc;
  u32x4 g0;
  g0[0] = 1u;                                   // count=1 (valid user D#)
  g0[1] = lds_off;                              // LDS byte address
  g0[2] = (unsigned)ga;                         // global_addr[31:0]
  g0[3] = (unsigned)(ga >> 32) | (2u << 30);    // global_addr[56:32] | type=2
  i32x8 g1;
  g1[0] = (int)((2u << 16) | (1u << 20) | (4u << 22)); // 4B elems, pad 1dw/32dw
  g1[1] = (int)(1024u << 16);                   // tensor_dim0 = 1024
  g1[2] = (int)(1024u << 16);                   // tensor_dim1 = 1024
  g1[3] = (int)(32u << 16);                     // tile_dim0  = 32
  g1[4] = 16;                                   // tile_dim1  = 16
  g1[5] = 1024;                                 // tensor_dim0_stride = 1024
  g1[6] = 0;
  g1[7] = 0;
  i32x4 z4 = {0, 0, 0, 0};
#if __clang_major__ >= 23
  i32x8 z8 = {0, 0, 0, 0, 0, 0, 0, 0};
  __builtin_amdgcn_tensor_load_to_lds(g0, g1, z4, z4, z8, 0);
#else
  __builtin_amdgcn_tensor_load_to_lds(g0, g1, z4, z4, 0);
#endif
}

// ---------------------------------------------------------------------------
// K3: fused masked-softmax aggregation. Block = (b, 16-row n-tile); wave h
// owns head h. Adjacency tiles arrive via double-buffered TDM.
// ---------------------------------------------------------------------------
__global__ __launch_bounds__(256) void gat_attn_wmma(
    const float* __restrict__ Adj, const __bf16* __restrict__ VT,
    const float* __restrict__ s_ws, const float* __restrict__ t_ws,
    float* __restrict__ out) {
  __shared__ float t_lds[Hh][Nn];        // 32 KB: per-head neighbor scores
  __shared__ float a_lds[2][16][33];     // double-buffered padded A tiles

  const int b    = blockIdx.y;
  const int n0   = blockIdx.x * 16;
  const int lane = threadIdx.x & 31;
  const int h    = threadIdx.x >> 5;
  const int r    = lane & 15;
  const int half = lane >> 4;

  // stage t[b,h,:] (read & written by this wave only -> no barrier needed)
  for (int i = lane; i < Nn; i += 32)
    t_lds[h][i] = t_ws[((size_t)b * Hh + h) * Nn + i];

  const float s_val = s_ws[((size_t)b * Hh + h) * Nn + n0 + r];
  const float* arow = Adj + ((size_t)b * Nn + n0) * Nn;

  // TDM prologue: tile 0 into buffer 0 (wave 0 drives the DMA engine)
  if (h == 0) tdm_load_a_tile(arow, (unsigned)(uintptr_t)&a_lds[0][0][0]);

  v8f acc0 = {}, acc1 = {};
  float zpart = 0.f;
  int buf = 0;

  for (int m0 = 0; m0 < Nn; m0 += 32) {
    if (h == 0) {
      if (m0 + 32 < Nn) {   // prefetch next tile into the other buffer
        tdm_load_a_tile(arow + m0 + 32,
                        (unsigned)(uintptr_t)&a_lds[buf ^ 1][0][0]);
        __builtin_amdgcn_s_wait_tensorcnt(1);  // current tile complete
      } else {
        __builtin_amdgcn_s_wait_tensorcnt(0);
      }
    }
    __syncthreads();   // tile visible to all heads

    // P tile in WMMA A layout; mask + leaky_relu + exp fused
    v16bf pa;
#pragma unroll
    for (int i = 0; i < 16; ++i) {
      const int k = kmapA(i, half);
      float sc = s_val + t_lds[h][m0 + k];
      sc = sc > 0.f ? sc : 0.01f * sc;                      // leaky_relu(0.01)
      const float p = (a_lds[buf][r][k] != 0.f) ? __expf(sc) : 0.f;
      zpart += p;
      pa[i] = (__bf16)p;
    }

    // V tiles from VT: 16 contiguous bf16 per operand -> 2 x b128 each
    const __bf16* vcol0 =
        VT + (((size_t)b * Hh + h) * Dd + r) * Nn + m0 + half * 16;
    const v16bf b0 = *(const v16bf*)vcol0;
    const v16bf b1 = *(const v16bf*)(vcol0 + (size_t)16 * Nn);

    acc0 = __builtin_amdgcn_wmma_f32_16x16x32_bf16(false, pa, false, b0,
                                                   (short)0, acc0, false, false);
    acc1 = __builtin_amdgcn_wmma_f32_16x16x32_bf16(false, pa, false, b1,
                                                   (short)0, acc1, false, false);

    __syncthreads();   // all heads done with buf before TDM refills it
    buf ^= 1;
  }

  // lanes L and L^16 hold complementary K-halves of row L%16
  const float ztot = zpart + __shfl_xor(zpart, 16, 32);

#pragma unroll
  for (int j = 0; j < 8; ++j) {
    const int row = j + half * 8;                 // C/D: VGPR j -> M=j / j+8
    const float inv = 1.f / __shfl(ztot, row, 32);
    const size_t o = ((size_t)b * Nn + n0 + row) * OUTD + h * Dd;
    const float o0 = acc0[j] * inv;
    const float o1 = acc1[j] * inv;
    out[o + r]      = o0 > 0.f ? o0 : 0.f;        // final relu
    out[o + 16 + r] = o1 > 0.f ? o1 : 0.f;
  }
}

// ---------------------------------------------------------------------------
extern "C" void kernel_launch(void* const* d_in, const int* in_sizes, int n_in,
                              void* d_out, int out_size, void* d_ws,
                              size_t ws_size, hipStream_t stream) {
  (void)in_sizes; (void)n_in; (void)out_size; (void)ws_size;

  const float* A   = (const float*)d_in[0];  // [8,1024,1024] f32
  const float* X   = (const float*)d_in[1];  // [8,1024,256]  f32
  const float* W   = (const float*)d_in[2];  // [256,256]     f32
  const float* fc1 = (const float*)d_in[3];  // [8,32]        f32
  const float* fc2 = (const float*)d_in[4];  // [8,32]        f32
  float* out = (float*)d_out;                // [8,1024,256]  f32

  // workspace layout (~4.85 MB)
  __bf16* VT = (__bf16*)d_ws;                                  // 4 MB
  char* p = (char*)d_ws + (size_t)Bn * Hh * Dd * Nn * 2;
  __bf16* Wt = (__bf16*)p;          p += (size_t)IND * OUTD * 2;  // 128 KB
  float* s_ws = (float*)p;          p += (size_t)Bn * Hh * Nn * 4;
  float* t_ws = (float*)p;

  gat_wt<<<(IND * OUTD) / 256, 256, 0, stream>>>(W, Wt);
  gat_proj_wmma<<<1024, 256, 0, stream>>>(X, Wt, VT);
  gat_scores<<<Bn * Hh, 256, 0, stream>>>(VT, fc1, fc2, s_ws, t_ws);
  gat_attn_wmma<<<dim3(Nn / 16, Bn), 256, 0, stream>>>(A, VT, s_ws, t_ws, out);
}
